// roundNet_4578435138046
// MI455X (gfx1250) — matrix operand order repaired and checked by reading, not compiled, not run on hardware
//
#include <hip/hip_runtime.h>
#include <hip/hip_bf16.h>

// ---------------------------------------------------------------------------
// CDNA5 (gfx1250) fused LSTM trajectory net.
//   - Encoder/decoder LSTMs run fully fused per 64-row tile, wave32, WMMA
//     v_wmma_f32_16x16x32_f16, c-state resident in accumulator-layout VGPRs,
//     h ping-ponged in LDS (f16), weights resident in LDS (f16, [K][N]).
//   - Decoder input is time-invariant -> xW precomputed once with WMMA and
//     cached in LDS in fragment layout.
//   - Gate activations use v_exp_f32 + v_rcp_f32 (no IEEE divide sequences)
//     to shorten the serial VALU tail on the recurrence critical path.
// ---------------------------------------------------------------------------

typedef __attribute__((ext_vector_type(16))) _Float16 v16h;
typedef __attribute__((ext_vector_type(8)))  _Float16 v8h;
typedef __attribute__((ext_vector_type(8)))  float    v8f;
typedef __attribute__((ext_vector_type(4)))  float    v4f;

#define DEV static __device__ __forceinline__

DEV float rcp_(float x) { return __builtin_amdgcn_rcpf(x); }       // v_rcp_f32
DEV float sigmoid_(float x) { return rcp_(1.0f + __expf(-x)); }
DEV float tanh_(float x) {
  float a = fabsf(x);
  float e = __expf(-2.0f * a);
  float t = (1.0f - e) * rcp_(1.0f + e);
  return copysignf(t, x);
}
DEV float lrelu_(float x) { return x > 0.0f ? x : 0.1f * x; }

DEV v8f zero8() {
  v8f z;
#pragma unroll
  for (int i = 0; i < 8; ++i) z[i] = 0.0f;
  return z;
}

// A operand (16xK tile, f16, 16x16x32 WMMA chunk starting at k0).
// ISA 7.12.2: lanes 0-15 -> M=lane, K in {k0..k0+7, k0+16..k0+23};
//             lanes 16-31 -> M=lane-16, K in {k0+8.., k0+24..}.
DEV v16h loadA(const _Float16* base, int stride, int lane, int k0) {
  const int m   = lane & 15;
  const int sel = (lane & 16) ? 8 : 0;
  const _Float16* p = base + m * stride + k0 + sel;
  v8h lo = *(const v8h*)(p);
  v8h hi = *(const v8h*)(p + 16);
  v16h a;
#pragma unroll
  for (int i = 0; i < 8; ++i) { a[i] = lo[i]; a[i + 8] = hi[i]; }
  return a;
}

// B operand (32x16 KxN tile, f16). Lane = N col (n0+lane%16), 16 halves = K in A's order.
// Weights stored in LDS as [k][n] with padded stride -> one row per lane.
DEV v16h loadB(const _Float16* w, int stride, int lane, int k0, int n0) {
  const int sel = (lane & 16) ? 8 : 0;
  const _Float16* p = w + (k0 + sel) * stride + n0 + (lane & 15);
  v16h b;
#pragma unroll
  for (int i = 0; i < 8; ++i) { b[i] = p[i * stride]; b[i + 8] = p[(i + 16) * stride]; }
  return b;
}

#define WMMA_F16(A, B, C) \
  __builtin_amdgcn_wmma_f32_16x16x32_f16(false, (A), false, (B), (short)0, (C), false, false)

// ---------------------------------------------------------------------------
// Encoder LSTM: x (16, rows, 2) -> embed(32) -> LSTM(64), final h to global.
// 64 rows / block, 256 threads (8 waves). 192 WMMAs per block per timestep.
// ---------------------------------------------------------------------------
#define E_WIH   0          // f16 [32][264]
#define E_WHH   16896      // f16 [64][264]
#define E_SX    50688      // f16 [64][40]
#define E_SH    55808      // f16 [2][64][72]
#define E_BIAS  74240      // f32 [256]
#define E_WIP   75264      // f32 [64]
#define E_BIP   75520      // f32 [32]
#define ENC_SMEM 75648

__global__ __launch_bounds__(256) void encoder_lstm(
    const float* __restrict__ x, int rows,
    const float* __restrict__ W_ip, const float* __restrict__ b_ip,
    const float* __restrict__ W_ih, const float* __restrict__ W_hh,
    const float* __restrict__ b_ih, const float* __restrict__ b_hh,
    float* __restrict__ h_out)
{
  extern __shared__ char smem[];
  _Float16* sWih = (_Float16*)(smem + E_WIH);
  _Float16* sWhh = (_Float16*)(smem + E_WHH);
  _Float16* sX   = (_Float16*)(smem + E_SX);
  _Float16* sH   = (_Float16*)(smem + E_SH);
  float*    sBias= (float*)(smem + E_BIAS);
  float*    sWip = (float*)(smem + E_WIP);
  float*    sBip = (float*)(smem + E_BIP);

  const int tid = threadIdx.x, lane = tid & 31, wave = tid >> 5;
  const int rowBase = blockIdx.x * 64;

  // Weights -> LDS f16, transposed to [k][gate] (WMMA-B friendly).
  for (int i = tid; i < 256 * 32; i += 256) {
    int g = i >> 5, k = i & 31;
    sWih[k * 264 + g] = (_Float16)W_ih[i];
  }
  for (int i = tid; i < 256 * 64; i += 256) {
    int g = i >> 6, k = i & 63;
    sWhh[k * 264 + g] = (_Float16)W_hh[i];
  }
  sBias[tid] = b_ih[tid] + b_hh[tid];
  if (tid < 64) sWip[tid] = W_ip[tid];
  if (tid < 32) sBip[tid] = b_ip[tid];
  for (int i = tid; i < 64 * 72; i += 256) sH[i] = (_Float16)0.0f;  // h0 = 0
  __syncthreads();

  // Each wave owns 2 (row-tile, enc-col-tile) pairs; c-state stays in VGPRs.
  const int p0 = wave * 2;
  v8f cst[2];
  cst[0] = zero8(); cst[1] = zero8();

  const int embR  = tid >> 2;        // 0..63
  const int embC0 = (tid & 3) * 8;   // 0,8,16,24

#pragma unroll 1
  for (int t = 0; t < 16; ++t) {
    // Phase 1: embedding emb = lrelu(x @ W_ip^T + b_ip), f16 into LDS.
    {
      float x0 = 0.0f, x1 = 0.0f;
      int gr = rowBase + embR;
      if (gr < rows) {
        const float* xp = x + ((size_t)t * rows + gr) * 2;
        x0 = xp[0]; x1 = xp[1];
      }
#pragma unroll
      for (int cc = 0; cc < 8; ++cc) {
        int c = embC0 + cc;
        float e = lrelu_(x0 * sWip[2 * c] + x1 * sWip[2 * c + 1] + sBip[c]);
        sX[embR * 40 + c] = (_Float16)e;
      }
    }
    __syncthreads();

    const int rbuf = t & 1;
    const _Float16* sHr = sH + rbuf * (64 * 72);
    _Float16*       sHw = sH + (1 - rbuf) * (64 * 72);

#pragma unroll
    for (int pp = 0; pp < 2; ++pp) {
      const int p = p0 + pp, rt = p & 3, et = p >> 2;
      const _Float16* aXb = sX  + rt * 16 * 40;
      const _Float16* aHb = sHr + rt * 16 * 72;
      v16h ax  = loadA(aXb, 40, lane, 0);
      v16h ah0 = loadA(aHb, 72, lane, 0);
      v16h ah1 = loadA(aHb, 72, lane, 32);
      v8f g4[4];
#pragma unroll
      for (int q = 0; q < 4; ++q) {
        const int n0 = q * 64 + et * 16;
        v8f acc = zero8();
        acc = WMMA_F16(ax,  loadB(sWih, 264, lane, 0,  n0), acc);
        acc = WMMA_F16(ah0, loadB(sWhh, 264, lane, 0,  n0), acc);
        acc = WMMA_F16(ah1, loadB(sWhh, 264, lane, 32, n0), acc);
        g4[q] = acc;
      }
      const int nc = et * 16 + (lane & 15);
      const float bI = sBias[nc], bF = sBias[64 + nc];
      const float bG = sBias[128 + nc], bO = sBias[192 + nc];
      const int rb = rt * 16 + ((lane & 16) ? 8 : 0);
#pragma unroll
      for (int j = 0; j < 8; ++j) {
        float iv = sigmoid_(g4[0][j] + bI);
        float fv = sigmoid_(g4[1][j] + bF);
        float gv = tanh_(g4[2][j] + bG);
        float ov = sigmoid_(g4[3][j] + bO);
        float cn = fv * cst[pp][j] + iv * gv;
        cst[pp][j] = cn;
        float hv = ov * tanh_(cn);
        int rl = rb + j;
        sHw[rl * 72 + nc] = (_Float16)hv;
        if (t == 15) {
          int gr = rowBase + rl;
          if (gr < rows) h_out[(size_t)gr * 64 + nc] = hv;
        }
      }
    }
    __syncthreads();
  }
}

// ---------------------------------------------------------------------------
// Decoder LSTM: input constant over the 25 steps -> precompute xW once.
// 64 rows / block; W_hh_d^T + xW fragments + h ping-pong all in LDS (~297KB).
// 512 WMMAs per block per timestep.
// ---------------------------------------------------------------------------
#define D_SW    0          // f16 [128][520] (W_ih_d^T first, then W_hh_d^T)
#define D_SH    133120     // f16 [2][64][136]
#define D_SXW   167936     // f32 [128 tiles][32 lanes][8]
#define D_BIAS  299008     // f32 [512]
#define D_WOP   301056     // f32 [5][128]
#define D_BOP   303616     // f32 [5]
#define DEC_SMEM 303648

__global__ __launch_bounds__(256) void decoder_lstm(
    const float* __restrict__ enc_full,
    const float* __restrict__ W_ih, const float* __restrict__ W_hh,
    const float* __restrict__ b_ih, const float* __restrict__ b_hh,
    const float* __restrict__ W_op, const float* __restrict__ b_op,
    float* __restrict__ out)
{
  extern __shared__ char smem[];
  _Float16* sW   = (_Float16*)(smem + D_SW);
  _Float16* sH   = (_Float16*)(smem + D_SH);
  float*    sXW  = (float*)(smem + D_SXW);
  float*    sBias= (float*)(smem + D_BIAS);
  float*    sWop = (float*)(smem + D_WOP);
  float*    sBop = (float*)(smem + D_BOP);

  const int tid = threadIdx.x, lane = tid & 31, wave = tid >> 5;
  const int rowBase = blockIdx.x * 64;

  // W_ih_d^T (K padded 101->128 with zeros) into sW; enc tile (f16, zero pad)
  // into sH buffer 0 (reused as A operand for the xW GEMM).
  for (int i = tid; i < 128 * 512; i += 256) {
    int k = i >> 9, g = i & 511;
    float v = (k < 101) ? W_ih[(size_t)g * 101 + k] : 0.0f;
    sW[k * 520 + g] = (_Float16)v;
  }
  for (int i = tid; i < 64 * 136; i += 256) {
    int r = i / 136, c = i - r * 136;
    float v = (c < 101) ? enc_full[(size_t)(rowBase + r) * 101 + c] : 0.0f;
    sH[r * 136 + c] = (_Float16)v;
  }
  for (int i = tid; i < 512; i += 256) sBias[i] = b_ih[i] + b_hh[i];
  for (int i = tid; i < 640; i += 256) sWop[i] = W_op[i];
  if (tid < 5) sBop[tid] = b_op[tid];
  __syncthreads();

  // Phase 0: xW = enc_full @ W_ih_d^T + (b_ih+b_hh), stored as WMMA fragments.
  const int p0 = wave * 4;
#pragma unroll
  for (int pp = 0; pp < 4; ++pp) {
    const int p = p0 + pp, rt = p & 3, et = p >> 2;
    const _Float16* aB = sH + rt * 16 * 136;
    v16h a0 = loadA(aB, 136, lane, 0);
    v16h a1 = loadA(aB, 136, lane, 32);
    v16h a2 = loadA(aB, 136, lane, 64);
    v16h a3 = loadA(aB, 136, lane, 96);
#pragma unroll
    for (int q = 0; q < 4; ++q) {
      const int n0 = q * 128 + et * 16;
      const float bv = sBias[n0 + (lane & 15)];
      v8f acc;
#pragma unroll
      for (int j = 0; j < 8; ++j) acc[j] = bv;
      acc = WMMA_F16(a0, loadB(sW, 520, lane, 0,  n0), acc);
      acc = WMMA_F16(a1, loadB(sW, 520, lane, 32, n0), acc);
      acc = WMMA_F16(a2, loadB(sW, 520, lane, 64, n0), acc);
      acc = WMMA_F16(a3, loadB(sW, 520, lane, 96, n0), acc);
      float* fp = sXW + ((size_t)(p * 4 + q) * 32 + lane) * 8;
      v4f lo, hi;
#pragma unroll
      for (int j = 0; j < 4; ++j) { lo[j] = acc[j]; hi[j] = acc[j + 4]; }
      *(v4f*)(fp) = lo;
      *(v4f*)(fp + 4) = hi;
    }
  }
  __syncthreads();

  // Swap sW to W_hh_d^T; zero h0 buffer.
  for (int i = tid; i < 128 * 512; i += 256) {
    int k = i >> 9, g = i & 511;
    sW[k * 520 + g] = (_Float16)W_hh[(size_t)g * 128 + k];
  }
  for (int i = tid; i < 64 * 136; i += 256) sH[i] = (_Float16)0.0f;
  __syncthreads();

  v8f cst[4];
#pragma unroll
  for (int pp = 0; pp < 4; ++pp) cst[pp] = zero8();

#pragma unroll 1
  for (int t = 0; t < 25; ++t) {
    const int rbuf = t & 1;
    const _Float16* sHr = sH + rbuf * (64 * 136);
    _Float16*       sHw = sH + (1 - rbuf) * (64 * 136);

#pragma unroll
    for (int pp = 0; pp < 4; ++pp) {
      const int p = p0 + pp, rt = p & 3, et = p >> 2;
      const _Float16* aB = sHr + rt * 16 * 136;
      v16h a0 = loadA(aB, 136, lane, 0);
      v16h a1 = loadA(aB, 136, lane, 32);
      v16h a2 = loadA(aB, 136, lane, 64);
      v16h a3 = loadA(aB, 136, lane, 96);
      v8f g4[4];
#pragma unroll
      for (int q = 0; q < 4; ++q) {
        const int n0 = q * 128 + et * 16;
        const float* fp = sXW + ((size_t)(p * 4 + q) * 32 + lane) * 8;
        v4f lo = *(const v4f*)(fp);
        v4f hi = *(const v4f*)(fp + 4);
        v8f acc;
#pragma unroll
        for (int j = 0; j < 4; ++j) { acc[j] = lo[j]; acc[j + 4] = hi[j]; }
        acc = WMMA_F16(a0, loadB(sW, 520, lane, 0,  n0), acc);
        acc = WMMA_F16(a1, loadB(sW, 520, lane, 32, n0), acc);
        acc = WMMA_F16(a2, loadB(sW, 520, lane, 64, n0), acc);
        acc = WMMA_F16(a3, loadB(sW, 520, lane, 96, n0), acc);
        g4[q] = acc;
      }
      const int nc = et * 16 + (lane & 15);
      const int rb = rt * 16 + ((lane & 16) ? 8 : 0);
#pragma unroll
      for (int j = 0; j < 8; ++j) {
        float iv = sigmoid_(g4[0][j]);
        float fv = sigmoid_(g4[1][j]);
        float gv = tanh_(g4[2][j]);
        float ov = sigmoid_(g4[3][j]);
        float cn = fv * cst[pp][j] + iv * gv;
        cst[pp][j] = cn;
        float hv = ov * tanh_(cn);
        sHw[(rb + j) * 136 + nc] = (_Float16)hv;
      }
    }
    __syncthreads();

    // Output projection + activation: fut[t] = act(h @ W_op^T + b_op).
    if (tid < 320) {
      int r = tid / 5, o = tid - r * 5;
      const _Float16* hp = sHw + r * 136;
      const float* wp = sWop + o * 128;
      float s = sBop[o];
#pragma unroll 8
      for (int k = 0; k < 128; ++k) s += (float)hp[k] * wp[k];
      if (o == 2 || o == 3) s = __expf(s);
      else if (o == 4) s = tanh_(s);
      out[(size_t)t * 40960 + (size_t)(rowBase + r) * 5 + o] = s;
    }
    __syncthreads();
  }
}

// ---------------------------------------------------------------------------
// Small glue kernels
// ---------------------------------------------------------------------------
__global__ void scan_counts_kernel(const int* __restrict__ counts,
                                   int* __restrict__ offsets)
{
  __shared__ int part[256];
  int tid = threadIdx.x, base = tid * 32, s = 0;
  for (int i = 0; i < 32; ++i) s += counts[base + i];
  part[tid] = s;
  __syncthreads();
  if (tid == 0) {
    int acc = 0;
    for (int i = 0; i < 256; ++i) { int v = part[i]; part[i] = acc; acc += v; }
  }
  __syncthreads();
  int run = part[tid];
  for (int i = 0; i < 32; ++i) { offsets[base + i] = run; run += counts[base + i]; }
}

__global__ __launch_bounds__(256) void dyn_onehot_kernel(
    const float* __restrict__ h_hist,
    const float* __restrict__ W_dyn, const float* __restrict__ b_dyn,
    const float* __restrict__ lat_enc, const float* __restrict__ lon_enc,
    float* __restrict__ enc_full)
{
  int tid = threadIdx.x;
  int row = blockIdx.x * 8 + (tid >> 5);
  int col = tid & 31;
  const float* hp = h_hist + (size_t)row * 64;
  const float* wp = W_dyn + col * 64;
  float s = b_dyn[col];
#pragma unroll 8
  for (int k = 0; k < 64; ++k) s += hp[k] * wp[k];
  enc_full[(size_t)row * 101 + 64 + col] = lrelu_(s);
  if (col < 3)      enc_full[(size_t)row * 101 + 96 + col] = lat_enc[row * 3 + col];
  else if (col < 5) enc_full[(size_t)row * 101 + 96 + col] = lon_enc[row * 2 + (col - 3)];
}

__global__ __launch_bounds__(256) void pool_kernel(
    const float* __restrict__ nbrs, int N, const float* __restrict__ nbrs_enc,
    const float* __restrict__ W_rel, const float* __restrict__ b_rel,
    const float* __restrict__ W_pp,  const float* __restrict__ b_pp,
    float* __restrict__ pool_h)
{
  int idx = blockIdx.x * 256 + threadIdx.x;
  int row = idx >> 6, col = idx & 63;
  if (row >= N) return;
  const float* lp = nbrs + ((size_t)15 * N + row) * 2;
  float x0 = lp[0], x1 = lp[1];
  const float* wpa = W_pp + col * 128;
  const float* ep  = nbrs_enc + (size_t)row * 64;
  float s = b_pp[col];
#pragma unroll 4
  for (int e = 0; e < 64; ++e) {
    float re = W_rel[e * 2] * x0 + W_rel[e * 2 + 1] * x1 + b_rel[e];
    s += wpa[e] * re + wpa[64 + e] * ep[e];
  }
  pool_h[(size_t)row * 64 + col] = fmaxf(s, 0.0f);
}

__global__ void soc_max_kernel(const float* __restrict__ pool_h,
                               const int* __restrict__ offsets,
                               const int* __restrict__ counts,
                               float* __restrict__ enc_full)
{
  int b = blockIdx.x, c = threadIdx.x;   // 64 threads = 64 cols
  int off = offsets[b], cnt = counts[b];
  float m = 0.0f;  // values are post-ReLU (>=0), so 0 is the correct identity
  for (int i = 0; i < cnt; ++i) m = fmaxf(m, pool_h[((size_t)(off + i)) * 64 + c]);
  enc_full[(size_t)b * 101 + c] = m;
}

__global__ __launch_bounds__(256) void latlon_kernel(
    const float* __restrict__ enc_full,
    const float* __restrict__ W_lat, const float* __restrict__ b_lat,
    const float* __restrict__ W_lon, const float* __restrict__ b_lon,
    float* __restrict__ out)
{
  int row = blockIdx.x * 256 + threadIdx.x;
  const float* ep = enc_full + (size_t)row * 101;
  float l0 = b_lat[0], l1 = b_lat[1], l2 = b_lat[2];
  float n0 = b_lon[0], n1 = b_lon[1];
#pragma unroll 4
  for (int k = 0; k < 96; ++k) {
    float e = ep[k];
    l0 += e * W_lat[k]; l1 += e * W_lat[96 + k]; l2 += e * W_lat[192 + k];
    n0 += e * W_lon[k]; n1 += e * W_lon[96 + k];
  }
  float m = fmaxf(l0, fmaxf(l1, l2));
  float e0 = __expf(l0 - m), e1 = __expf(l1 - m), e2 = __expf(l2 - m);
  float inv = rcp_(e0 + e1 + e2);
  out[1024000 + (size_t)row * 3 + 0] = e0 * inv;
  out[1024000 + (size_t)row * 3 + 1] = e1 * inv;
  out[1024000 + (size_t)row * 3 + 2] = e2 * inv;
  float mn = fmaxf(n0, n1);
  float f0 = __expf(n0 - mn), f1 = __expf(n1 - mn);
  float invn = rcp_(f0 + f1);
  out[1048576 + (size_t)row * 2 + 0] = f0 * invn;
  out[1048576 + (size_t)row * 2 + 1] = f1 * invn;
}

// ---------------------------------------------------------------------------
extern "C" void kernel_launch(void* const* d_in, const int* in_sizes, int n_in,
                              void* d_out, int out_size, void* d_ws, size_t ws_size,
                              hipStream_t stream)
{
  const float* hist    = (const float*)d_in[0];
  const float* nbrs    = (const float*)d_in[1];
  const int*   counts  = (const int*)  d_in[2];
  const float* lat_enc = (const float*)d_in[3];
  const float* lon_enc = (const float*)d_in[4];
  const float* W_ip    = (const float*)d_in[6];
  const float* b_ip    = (const float*)d_in[7];
  const float* W_ih_e  = (const float*)d_in[8];
  const float* W_hh_e  = (const float*)d_in[9];
  const float* b_ih_e  = (const float*)d_in[10];
  const float* b_hh_e  = (const float*)d_in[11];
  const float* W_dyn   = (const float*)d_in[12];
  const float* b_dyn   = (const float*)d_in[13];
  const float* W_rel   = (const float*)d_in[14];
  const float* b_rel   = (const float*)d_in[15];
  const float* W_pp    = (const float*)d_in[16];
  const float* b_pp    = (const float*)d_in[17];
  const float* W_lat   = (const float*)d_in[18];
  const float* b_lat   = (const float*)d_in[19];
  const float* W_lon   = (const float*)d_in[20];
  const float* b_lon   = (const float*)d_in[21];
  const float* W_ih_d  = (const float*)d_in[22];
  const float* W_hh_d  = (const float*)d_in[23];
  const float* b_ih_d  = (const float*)d_in[24];
  const float* b_hh_d  = (const float*)d_in[25];
  const float* W_op    = (const float*)d_in[26];
  const float* b_op    = (const float*)d_in[27];

  const int N = in_sizes[1] / (16 * 2);   // nbrs is (16, N, 2)
  float* out = (float*)d_out;

  char* ws = (char*)d_ws;
  size_t off = 0;
  auto carve = [&](size_t bytes) -> void* {
    void* p = ws + off;
    off = (off + bytes + 255) & ~(size_t)255;
    return p;
  };
  int*   offsets  = (int*)  carve((size_t)8192 * sizeof(int));
  float* h_hist   = (float*)carve((size_t)8192 * 64 * sizeof(float));
  float* enc_full = (float*)carve((size_t)8192 * 101 * sizeof(float));
  float* nbrs_enc = (float*)carve((size_t)(N > 0 ? N : 1) * 64 * sizeof(float));
  float* pool_h   = (float*)carve((size_t)(N > 0 ? N : 1) * 64 * sizeof(float));

  scan_counts_kernel<<<1, 256, 0, stream>>>(counts, offsets);

  encoder_lstm<<<8192 / 64, 256, ENC_SMEM, stream>>>(
      hist, 8192, W_ip, b_ip, W_ih_e, W_hh_e, b_ih_e, b_hh_e, h_hist);

  if (N > 0) {
    encoder_lstm<<<(N + 63) / 64, 256, ENC_SMEM, stream>>>(
        nbrs, N, W_ip, b_ip, W_ih_e, W_hh_e, b_ih_e, b_hh_e, nbrs_enc);
    pool_kernel<<<(int)(((size_t)N * 64 + 255) / 256), 256, 0, stream>>>(
        nbrs, N, nbrs_enc, W_rel, b_rel, W_pp, b_pp, pool_h);
  }

  dyn_onehot_kernel<<<8192 / 8, 256, 0, stream>>>(
      h_hist, W_dyn, b_dyn, lat_enc, lon_enc, enc_full);

  soc_max_kernel<<<8192, 64, 0, stream>>>(pool_h, offsets, counts, enc_full);

  latlon_kernel<<<8192 / 256, 256, 0, stream>>>(
      enc_full, W_lat, b_lat, W_lon, b_lon, out);

  decoder_lstm<<<8192 / 64, 256, DEC_SMEM, stream>>>(
      enc_full, W_ih_d, W_hh_d, b_ih_d, b_hh_d, W_op, b_op, out);
}
